// SoftAttentionLayer_51462298141140
// MI455X (gfx1250) — matrix operand order repaired and checked by reading, not compiled, not run on hardware
//
#include <hip/hip_runtime.h>
#include <hip/hip_bf16.h>

// ---------------------------------------------------------------------------
// Shapes (hard-coded from reference): B=2, N=512, D=64, H=4, HD=16, CD=192
// ---------------------------------------------------------------------------
#define NB 2
#define NN 512
#define ND 64
#define NCD 192

typedef __attribute__((ext_vector_type(16))) _Float16 v16h;
typedef __attribute__((ext_vector_type(8)))  float    v8f;

// ---------------- workspace layout (byte offsets) --------------------------
#define WS_GE    0                                   // f32 [B*N*D]   graph_embs
#define WS_QKV   (WS_GE   + NB*NN*ND*4)              // f32 [B*N*192] qkv
#define WS_ATT   (WS_QKV  + NB*NN*NCD*4)             // f32 [B*N*D]   attn_embs
#define WS_W1WT  (WS_ATT  + NB*NN*ND*4)              // f16 [192][192]
#define WS_W1VT  (WS_W1WT + NCD*NCD*2)               // f16 [192][192]
#define WS_W2VT  (WS_W1VT + NCD*NCD*2)               // f16 [64][192]
#define WS_WE1T  (WS_W2VT + ND*NCD*2)                // f16 [128][128]
#define WS_WE2T  (WS_WE1T + 128*128*2)               // f16 [64][128]
#define WS_END   (WS_WE2T + ND*128*2)

__device__ __forceinline__ float leaky01(float x) { return x > 0.f ? x : 0.1f * x; }

// ---------------------------------------------------------------------------
// WMMA fragment loaders (contiguous k order, same for A and B, wave32)
// A 16x32: lane&15 = M, lane>>4 selects K half (+16); VGPR r: K=2r
// ---------------------------------------------------------------------------
__device__ __forceinline__ v16h frag_a_lds(const _Float16* base, int ld, int m0, int k0) {
  const int lane = threadIdx.x & 31;
  const _Float16* row = base + (m0 + (lane & 15)) * ld + k0 + ((lane >> 4) << 4);
  v16h out;
#pragma unroll
  for (int r = 0; r < 8; ++r) {
    const int k = 2 * r;
    out[2 * r]     = row[k];
    out[2 * r + 1] = row[k + 1];
  }
  return out;
}

// B 32x16 (column-major): lane&15 = N, lane>>4 selects K half (+16);
// with WT stored as [N][K] f16, each lane reads 16 contiguous halfs (32B).
__device__ __forceinline__ v16h frag_b_glb(const _Float16* WT, int ldk, int n0, int k0) {
  const int lane = threadIdx.x & 31;
  const _Float16* row = WT + (n0 + (lane & 15)) * ldk + k0 + ((lane >> 4) << 4);
  return *(const v16h*)row;
}

// C/D 16x16 f32: VGPR r -> m = r + 8*(lane>=16), n = lane&15
__device__ __forceinline__ void store_frag_leaky_f16(_Float16* dst, int ld, int m0, int n0,
                                                     v8f acc, const float* __restrict__ bias) {
  const int lane = threadIdx.x & 31;
  const int n  = n0 + (lane & 15);
  const int mb = m0 + ((lane >> 4) << 3);
  const float bs = bias[n];
#pragma unroll
  for (int r = 0; r < 8; ++r)
    dst[(mb + r) * ld + n] = (_Float16)leaky01(acc[r] + bs);
}

// ---------------------------------------------------------------------------
// k0: convert + transpose small weights to f16 [N][K]
// ---------------------------------------------------------------------------
__global__ void k0_prep(const float* __restrict__ W1w, const float* __restrict__ W1v,
                        const float* __restrict__ W2v, const float* __restrict__ We1,
                        const float* __restrict__ We2,
                        _Float16* __restrict__ w1wt, _Float16* __restrict__ w1vt,
                        _Float16* __restrict__ w2vt, _Float16* __restrict__ we1t,
                        _Float16* __restrict__ we2t) {
  int idx = blockIdx.x * blockDim.x + threadIdx.x;
  if (idx < NCD * NCD) { int n = idx / NCD, k = idx % NCD; w1wt[idx] = (_Float16)W1w[k * NCD + n]; return; }
  idx -= NCD * NCD;
  if (idx < NCD * NCD) { int n = idx / NCD, k = idx % NCD; w1vt[idx] = (_Float16)W1v[k * NCD + n]; return; }
  idx -= NCD * NCD;
  if (idx < ND * NCD)  { int n = idx / NCD, k = idx % NCD; w2vt[idx] = (_Float16)W2v[k * ND + n]; return; }
  idx -= ND * NCD;
  if (idx < 128 * 128) { int n = idx / 128, k = idx % 128; we1t[idx] = (_Float16)We1[k * 128 + n]; return; }
  idx -= 128 * 128;
  if (idx < ND * 128)  { int n = idx / 128, k = idx % 128; we2t[idx] = (_Float16)We2[k * ND + n]; return; }
}

// ---------------------------------------------------------------------------
// k1: per (b,i) -> pair towers (WMMA), softmax over j, weighted aggregation
// ---------------------------------------------------------------------------
#define LD1 200   // padded leading dim (halfs) for 192-wide tiles
#define VLD 72    // padded leading dim for the 512x64 value buffer

__global__ __launch_bounds__(256)
void k1_pair_tower(const float* __restrict__ nodes, const float* __restrict__ edges,
                   const _Float16* __restrict__ W1wT, const float* __restrict__ b1w,
                   const float* __restrict__ W2w,  const float* __restrict__ b2w,
                   const _Float16* __restrict__ W1vT, const float* __restrict__ b1v,
                   const _Float16* __restrict__ W2vT, const float* __restrict__ b2v,
                   float* __restrict__ ge) {
  __shared__ _Float16 gf[64 * LD1];
  __shared__ _Float16 h1[64 * LD1];
  __shared__ _Float16 valsm[NN * VLD];
  __shared__ float    logitsS[NN];
  __shared__ float    red[256];

  const int tid  = threadIdx.x;
  const int wave = tid >> 5;
  const int lane = tid & 31;
  const int b = blockIdx.x >> 9;
  const int i = blockIdx.x & (NN - 1);
  const float* nrow_i = nodes + ((size_t)b * NN + i) * ND;
  const float* erow_i = edges + (((size_t)b * NN + i) * NN) * ND;

  for (int jc = 0; jc < 8; ++jc) {
    // ---- stage gf chunk: [nodes_i | nodes_j | edges_ij] as f16 ----
    for (int idx = tid; idx < 64 * NCD; idx += 256) {
      const int r = idx / NCD, c = idx % NCD;
      const int j = jc * 64 + r;
      float v;
      if (c < ND)            v = nrow_i[c];
      else if (c < 2 * ND)   v = nodes[((size_t)b * NN + j) * ND + (c - ND)];
      else                   v = erow_i[(size_t)j * ND + (c - 2 * ND)];
      gf[r * LD1 + c] = (_Float16)v;
    }
    // prefetch next chunk's edge rows (64 rows x 256B = 16KB) into cache while
    // the WMMA towers below execute (emits global_prefetch_b8)
    if (jc < 7) {
      const float* pf = erow_i + ((size_t)(jc + 1) * 64) * ND + tid * 16;
      __builtin_prefetch(pf, 0, 1);
    }
    __syncthreads();

    // ---- h1 = leaky(gf @ W1w + b1w) : M=64, N=192, K=192 ----
    for (int t = wave; t < 48; t += 8) {
      const int m0 = (t & 3) << 4, n0 = (t >> 2) << 4;
      v8f acc = {};
#pragma unroll
      for (int kk = 0; kk < 6; ++kk) {
        v16h a  = frag_a_lds(gf, LD1, m0, kk * 32);
        v16h bb = frag_b_glb(W1wT, NCD, n0, kk * 32);
        acc = __builtin_amdgcn_wmma_f32_16x16x32_f16(false, a, false, bb, (short)0, acc, false, false);
      }
      store_frag_leaky_f16(h1, LD1, m0, n0, acc, b1w);
    }
    __syncthreads();

    // ---- logits = h1 @ W2w + b2w (N=1, VALU dot; 4 threads per row) ----
    {
      const int rr = tid >> 2, q = tid & 3;
      float p = 0.f;
      const _Float16* hr = h1 + rr * LD1 + q * 48;
      for (int c = 0; c < 48; ++c) p += (float)hr[c] * W2w[q * 48 + c];
      red[tid] = p;
      __syncthreads();
      if (q == 0)
        logitsS[jc * 64 + rr] = red[tid] + red[tid + 1] + red[tid + 2] + red[tid + 3] + b2w[0];
      __syncthreads();
    }

    // ---- h1 = leaky(gf @ W1v + b1v) ----
    for (int t = wave; t < 48; t += 8) {
      const int m0 = (t & 3) << 4, n0 = (t >> 2) << 4;
      v8f acc = {};
#pragma unroll
      for (int kk = 0; kk < 6; ++kk) {
        v16h a  = frag_a_lds(gf, LD1, m0, kk * 32);
        v16h bb = frag_b_glb(W1vT, NCD, n0, kk * 32);
        acc = __builtin_amdgcn_wmma_f32_16x16x32_f16(false, a, false, bb, (short)0, acc, false, false);
      }
      store_frag_leaky_f16(h1, LD1, m0, n0, acc, b1v);
    }
    __syncthreads();

    // ---- vals = h1 @ W2v + b2v : M=64, N=64, K=192 ----
    for (int t = wave; t < 16; t += 8) {
      const int m0 = (t & 3) << 4, n0 = (t >> 2) << 4;
      v8f acc = {};
#pragma unroll
      for (int kk = 0; kk < 6; ++kk) {
        v16h a  = frag_a_lds(h1, LD1, m0, kk * 32);
        v16h bb = frag_b_glb(W2vT, NCD, n0, kk * 32);
        acc = __builtin_amdgcn_wmma_f32_16x16x32_f16(false, a, false, bb, (short)0, acc, false, false);
      }
      const int n  = n0 + (lane & 15);
      const int mb = jc * 64 + m0 + ((lane >> 4) << 3);
      const float bs = b2v[n];
#pragma unroll
      for (int r = 0; r < 8; ++r)
        valsm[(mb + r) * VLD + n] = (_Float16)(acc[r] + bs);
    }
    __syncthreads();
  }

  // ---- softmax over j (512) ----
  float mx = -1e30f;
  for (int j = tid; j < NN; j += 256) mx = fmaxf(mx, logitsS[j]);
  red[tid] = mx; __syncthreads();
  for (int off = 128; off > 0; off >>= 1) { if (tid < off) red[tid] = fmaxf(red[tid], red[tid + off]); __syncthreads(); }
  mx = red[0];
  __syncthreads();
  float s = 0.f;
  for (int j = tid; j < NN; j += 256) { float e = __expf(logitsS[j] - mx); logitsS[j] = e; s += e; }
  red[tid] = s; __syncthreads();
  for (int off = 128; off > 0; off >>= 1) { if (tid < off) red[tid] += red[tid + off]; __syncthreads(); }
  const float inv = 1.f / red[0];
  __syncthreads();

  // ---- graph_embs[b,i,d] = sum_j w_j * vals[j][d] ----
  {
    const int part = tid >> 6, d = tid & 63;
    float acc = 0.f;
    for (int j = part * 128; j < part * 128 + 128; ++j)
      acc += logitsS[j] * (float)valsm[j * VLD + d];
    red[tid] = acc; __syncthreads();
    if (tid < 64)
      ge[((size_t)b * NN + i) * ND + tid] =
          (red[tid] + red[tid + 64] + red[tid + 128] + red[tid + 192]) * inv;
  }
}

// ---------------------------------------------------------------------------
// k2a: qkv = graph_embs @ Wqkv + bqkv  (tiny, VALU)
// ---------------------------------------------------------------------------
__global__ void k2a_qkv(const float* __restrict__ ge, const float* __restrict__ Wqkv,
                        const float* __restrict__ bqkv, float* __restrict__ qkv) {
  const int idx = blockIdx.x * blockDim.x + threadIdx.x;
  if (idx >= NB * NN * NCD) return;
  const int c = idx % NCD;
  const int row = idx / NCD;
  const float* g = ge + (size_t)row * ND;
  float s = bqkv[c];
  for (int d = 0; d < ND; ++d) s += g[d] * Wqkv[d * NCD + c];
  qkv[idx] = s;
}

// ---------------------------------------------------------------------------
// k2b: MHSA (H=4, HD=16), two-pass softmax in LDS, per (b,i) block
// ---------------------------------------------------------------------------
__global__ __launch_bounds__(128)
void k2b_attn(const float* __restrict__ qkv, float* __restrict__ attn) {
  __shared__ float qrow[64];
  __shared__ float lg[4 * NN];
  __shared__ float red2[8 + 128];
  const int tid = threadIdx.x;
  const int b = blockIdx.x >> 9, i = blockIdx.x & (NN - 1);
  const int h = tid >> 5, lane = tid & 31;
  const float* qbase = qkv + ((size_t)b * NN + i) * NCD;
  if (tid < 64) qrow[tid] = qbase[(tid >> 4) * 48 + (tid & 15)];
  __syncthreads();

  for (int jb = 0; jb < 16; ++jb) {
    const int j = jb * 32 + lane;
    const float* kr = qkv + ((size_t)b * NN + j) * NCD + h * 48 + 16;
    float s = 0.f;
#pragma unroll
    for (int e = 0; e < 16; ++e) s += qrow[h * 16 + e] * kr[e];
    lg[h * NN + j] = s * 0.25f;   // 1/sqrt(16)
  }
  __syncthreads();
  if (lane == 0) {
    float m = -1e30f;
    for (int j = 0; j < NN; ++j) m = fmaxf(m, lg[h * NN + j]);
    red2[h] = m;
  }
  __syncthreads();
  const float mx = red2[h];
  float part = 0.f;
  for (int jb = 0; jb < 16; ++jb) {
    const int j = jb * 32 + lane;
    const float e = __expf(lg[h * NN + j] - mx);
    lg[h * NN + j] = e;
    part += e;
  }
  red2[8 + tid] = part;
  __syncthreads();
  if (lane == 0) {
    float sum = 0.f;
    for (int l = 0; l < 32; ++l) sum += red2[8 + h * 32 + l];
    red2[4 + h] = sum;
  }
  __syncthreads();
  if (tid < 64) {
    const int h2 = tid >> 4, e = tid & 15;
    const float inv = 1.f / red2[4 + h2];
    float acc = 0.f;
    for (int j = 0; j < NN; ++j)
      acc += lg[h2 * NN + j] * qkv[((size_t)b * NN + j) * NCD + h2 * 48 + 32 + e];
    attn[((size_t)b * NN + i) * ND + h2 * 16 + e] = acc * inv;
  }
}

// ---------------------------------------------------------------------------
// k3: new_nodes = leaky(ln(node_mlp(attn) + nodes))
// ---------------------------------------------------------------------------
__global__ __launch_bounds__(64)
void k3_nodes(const float* __restrict__ attn, const float* __restrict__ nodes,
              const float* __restrict__ Wn1, const float* __restrict__ bn1,
              const float* __restrict__ Wn2, const float* __restrict__ bn2,
              const float* __restrict__ ln_g, const float* __restrict__ ln_b,
              float* __restrict__ out_nodes) {
  __shared__ float a[64], hb[64], red3[64];
  const int tid = threadIdx.x;
  const size_t row = blockIdx.x;   // b*N + i
  a[tid] = attn[row * ND + tid];
  __syncthreads();
  float s = bn1[tid];
  for (int d = 0; d < ND; ++d) s += a[d] * Wn1[d * ND + tid];
  hb[tid] = leaky01(s);
  __syncthreads();
  float o = bn2[tid];
  for (int d = 0; d < ND; ++d) o += hb[d] * Wn2[d * ND + tid];
  const float x = o + nodes[row * ND + tid];
  red3[tid] = x; __syncthreads();
  for (int off = 32; off > 0; off >>= 1) { if (tid < off) red3[tid] += red3[tid + off]; __syncthreads(); }
  const float mean = red3[0] * (1.f / 64.f);
  __syncthreads();
  const float dx = x - mean;
  red3[tid] = dx * dx; __syncthreads();
  for (int off = 32; off > 0; off >>= 1) { if (tid < off) red3[tid] += red3[tid + off]; __syncthreads(); }
  const float var = red3[0] * (1.f / 64.f);
  const float y = dx * rsqrtf(var + 1e-5f) * ln_g[tid] + ln_b[tid];
  out_nodes[row * ND + tid] = leaky01(y);
}

// ---------------------------------------------------------------------------
// k4: per (b,i) -> edge MLP (WMMA) over [attn_i|attn_j], doubled residual, LN
// ---------------------------------------------------------------------------
#define LDE 136   // padded leading dim for 128-wide tiles
#define OLD 66

__global__ __launch_bounds__(256)
void k4_edges(const float* __restrict__ attn, const _Float16* __restrict__ We1T,
              const float* __restrict__ be1, const _Float16* __restrict__ We2T,
              const float* __restrict__ be2, const float* __restrict__ ln_g,
              const float* __restrict__ ln_b, float* __restrict__ out_edges) {
  __shared__ _Float16 et[64 * LDE];
  __shared__ _Float16 ht[64 * LDE];
  __shared__ float    ot[64 * OLD];
  __shared__ float    mrow[64], rsrow[64];
  __shared__ float    lgS[64], lbS[64];
  const int tid = threadIdx.x;
  const int wave = tid >> 5, lane = tid & 31;
  const int b = blockIdx.x >> 9, i = blockIdx.x & (NN - 1);
  const float* arow_i = attn + ((size_t)b * NN + i) * ND;
  if (tid < 64) { lgS[tid] = ln_g[tid]; lbS[tid] = ln_b[tid]; }

  for (int jc = 0; jc < 8; ++jc) {
    for (int idx = tid; idx < 64 * 128; idx += 256) {
      const int r = idx >> 7, c = idx & 127;
      const float v = (c < ND) ? arow_i[c]
                               : attn[((size_t)b * NN + (jc * 64 + r)) * ND + (c - ND)];
      et[r * LDE + c] = (_Float16)v;
    }
    __syncthreads();

    // h = leaky(e @ We1 + be1): M=64, N=128, K=128
    for (int t = wave; t < 32; t += 8) {
      const int m0 = (t & 3) << 4, n0 = (t >> 2) << 4;
      v8f acc = {};
#pragma unroll
      for (int kk = 0; kk < 4; ++kk) {
        v16h a  = frag_a_lds(et, LDE, m0, kk * 32);
        v16h bb = frag_b_glb(We1T, 128, n0, kk * 32);
        acc = __builtin_amdgcn_wmma_f32_16x16x32_f16(false, a, false, bb, (short)0, acc, false, false);
      }
      store_frag_leaky_f16(ht, LDE, m0, n0, acc, be1);
    }
    __syncthreads();

    // ne = h @ We2 + be2: M=64, N=64, K=128
    for (int t = wave; t < 16; t += 8) {
      const int m0 = (t & 3) << 4, n0 = (t >> 2) << 4;
      v8f acc = {};
#pragma unroll
      for (int kk = 0; kk < 4; ++kk) {
        v16h a  = frag_a_lds(ht, LDE, m0, kk * 32);
        v16h bb = frag_b_glb(We2T, 128, n0, kk * 32);
        acc = __builtin_amdgcn_wmma_f32_16x16x32_f16(false, a, false, bb, (short)0, acc, false, false);
      }
      const int n  = n0 + (lane & 15);
      const int mb = m0 + ((lane >> 4) << 3);
      const float bs = be2[n];
#pragma unroll
      for (int r = 0; r < 8; ++r) ot[(mb + r) * OLD + n] = acc[r] + bs;
    }
    __syncthreads();

    // per-row LN stats for x = 2*ne
    if (tid < 64) {
      const float* row = &ot[tid * OLD];
      float m = 0.f;
      for (int c = 0; c < ND; ++c) m += row[c];
      m = m * (2.f / 64.f);
      float var = 0.f;
      for (int c = 0; c < ND; ++c) { const float v = 2.f * row[c] - m; var += v * v; }
      var *= (1.f / 64.f);
      mrow[tid]  = m;
      rsrow[tid] = rsqrtf(var + 1e-5f);
    }
    __syncthreads();

    // coalesced store: consecutive threads -> consecutive columns
    {
      const size_t base = ((((size_t)b * NN + i) * NN) + (size_t)jc * 64) * ND;
      for (int idx = tid; idx < 64 * 64; idx += 256) {
        const int r = idx >> 6, c = idx & 63;
        const float y = (2.f * ot[r * OLD + c] - mrow[r]) * rsrow[r] * lgS[c] + lbS[c];
        out_edges[base + idx] = leaky01(y);
      }
    }
    __syncthreads();
  }
}

// ---------------------------------------------------------------------------
extern "C" void kernel_launch(void* const* d_in, const int* in_sizes, int n_in,
                              void* d_out, int out_size, void* d_ws, size_t ws_size,
                              hipStream_t stream) {
  const float* nodes = (const float*)d_in[0];
  const float* edges = (const float*)d_in[1];
  const float* W1w  = (const float*)d_in[2];
  const float* b1w  = (const float*)d_in[3];
  const float* W2w  = (const float*)d_in[4];
  const float* b2w  = (const float*)d_in[5];
  const float* W1v  = (const float*)d_in[6];
  const float* b1v  = (const float*)d_in[7];
  const float* W2v  = (const float*)d_in[8];
  const float* b2v  = (const float*)d_in[9];
  const float* Wqkv = (const float*)d_in[10];
  const float* bqkv = (const float*)d_in[11];
  const float* Wn1  = (const float*)d_in[12];
  const float* bn1  = (const float*)d_in[13];
  const float* Wn2  = (const float*)d_in[14];
  const float* bn2  = (const float*)d_in[15];
  const float* We1  = (const float*)d_in[16];
  const float* be1  = (const float*)d_in[17];
  const float* We2  = (const float*)d_in[18];
  const float* be2  = (const float*)d_in[19];
  const float* ln_g = (const float*)d_in[20];
  const float* ln_b = (const float*)d_in[21];

  char* ws = (char*)d_ws;
  float*    ge    = (float*)(ws + WS_GE);
  float*    qkv   = (float*)(ws + WS_QKV);
  float*    attn  = (float*)(ws + WS_ATT);
  _Float16* W1wT  = (_Float16*)(ws + WS_W1WT);
  _Float16* W1vT  = (_Float16*)(ws + WS_W1VT);
  _Float16* W2vT  = (_Float16*)(ws + WS_W2VT);
  _Float16* We1T  = (_Float16*)(ws + WS_WE1T);
  _Float16* We2T  = (_Float16*)(ws + WS_WE2T);

  float* out_nodes = (float*)d_out;
  float* out_edges = out_nodes + (size_t)NB * NN * ND;

  const int prep_elems = NCD*NCD + NCD*NCD + ND*NCD + 128*128 + ND*128;
  k0_prep<<<(prep_elems + 255) / 256, 256, 0, stream>>>(W1w, W1v, W2v, We1, We2,
                                                        W1wT, W1vT, W2vT, We1T, We2T);
  k1_pair_tower<<<NB * NN, 256, 0, stream>>>(nodes, edges, W1wT, b1w, W2w, b2w,
                                             W1vT, b1v, W2vT, b2v, ge);
  k2a_qkv<<<(NB * NN * NCD + 255) / 256, 256, 0, stream>>>(ge, Wqkv, bqkv, qkv);
  k2b_attn<<<NB * NN, 128, 0, stream>>>(qkv, attn);
  k3_nodes<<<NB * NN, 64, 0, stream>>>(attn, nodes, Wn1, bn1, Wn2, bn2, ln_g, ln_b, out_nodes);
  k4_edges<<<NB * NN, 256, 0, stream>>>(attn, We1T, be1, We2T, be2, ln_g, ln_b, out_edges);
}
